// Encoder_RT_7361573945838
// MI455X (gfx1250) — matrix operand order repaired and checked by reading, not compiled, not run on hardware
//
#include <hip/hip_runtime.h>
#include <hip/hip_bf16.h>

typedef __attribute__((ext_vector_type(16))) _Float16 v16h;
typedef __attribute__((ext_vector_type(8)))  _Float16 v8h;
typedef __attribute__((ext_vector_type(8)))  float    v8f;

#define BN_EPS 1e-8f
#define NB 64          // batches per workgroup (4 waves x 16)
#define TPB 128        // threads per block (4 wave32)
#define FO_PER_WG 8    // output frequency positions per workgroup

// ---------------------------------------------------------------------------
// Implicit-GEMM conv stage: D[Cout x 16n] = Wf16[Cout x K] * patch[K x 16n]
// K order: k = df*(2*Cin) + dt*Cin + ci  (df = freq tap, dt = time tap)
// dt==0 -> cached previous frame, dt==1 -> current frame.
// XNHWC: stage-1 current frame is the raw input with layout [B, F, 1, 2].
// Epilogue: folded eval-BN (scale/shift) + per-channel PReLU, fp32 stores.
// ---------------------------------------------------------------------------
template<int CIN, int COUT, int FIN, int FOUT, int STRIDE, int KF, bool XNHWC>
__global__ __launch_bounds__(TPB)
void stage_kernel(const float* __restrict__ xin,
                  const float* __restrict__ cachein,
                  const float* __restrict__ W,
                  const float* __restrict__ bias,
                  const float* __restrict__ gamma,
                  const float* __restrict__ beta,
                  const float* __restrict__ mean,
                  const float* __restrict__ var,
                  const float* __restrict__ alpha,
                  float* __restrict__ out0,
                  float* __restrict__ out1)
{
  constexpr int KTOT   = KF * 2 * CIN;
  constexpr int KSTEPS = (KTOT + 31) / 32;
  constexpr int KPAD   = KSTEPS * 32;
  constexpr int MT     = COUT / 16;

  extern __shared__ char smem[];
  _Float16* Wh = (_Float16*)smem;            // [COUT][KPAD] f16 weights
  _Float16* Bh = Wh + COUT * KPAD;           // [NB][KPAD]  f16 im2col patch
  float* sscale = (float*)(Bh + NB * KPAD);  // [COUT] folded BN scale
  float* sshift = sscale + COUT;             // [COUT] folded BN shift
  float* salpha = sshift + COUT;             // [COUT] PReLU slope

  const int    tid     = threadIdx.x;
  const size_t b0      = (size_t)blockIdx.x * NB;
  const int    fo_base = blockIdx.y * FO_PER_WG;

  // ---- load weights to LDS as f16 in GEMM K-order (zero-pad K to KPAD) ----
  for (int idx = tid; idx < COUT * KPAD; idx += TPB) {
    int m = idx / KPAD, k = idx % KPAD;
    float w = 0.f;
    if (k < KTOT) {
      int df = k / (2 * CIN);
      int r  = k % (2 * CIN);
      int dt = r / CIN;
      int ci = r % CIN;
      w = W[(((size_t)m * CIN + ci) * 2 + dt) * KF + df];
    }
    Wh[idx] = (_Float16)w;
  }
  // ---- fold conv-bias + eval BN into per-channel scale/shift ----
  for (int c = tid; c < COUT; c += TPB) {
    float inv = rsqrtf(var[c] + BN_EPS);
    float s   = gamma[c] * inv;
    sscale[c] = s;
    sshift[c] = (bias[c] - mean[c]) * s + beta[c];
    salpha[c] = alpha[c];
  }
  __syncthreads();

  const int lane = tid & 31;
  const int wv   = tid >> 5;
  const int l15  = lane & 15;   // N index (B-frag) / M index (A-frag)
  const int lhi  = lane >> 4;   // lane half selects K sub-block

  for (int it = 0; it < FO_PER_WG; ++it) {
    int fo = fo_base + it;
    if (fo >= FOUT) break;      // uniform per block

    // ---- gather im2col patch for this fo into LDS (f32 -> f16) ----
    for (int idx = tid; idx < NB * KPAD; idx += TPB) {
      int nb = idx / KPAD, k = idx % KPAD;
      float v = 0.f;
      if (k < KTOT) {
        int df = k / (2 * CIN);
        int r  = k % (2 * CIN);
        int dt = r / CIN;
        int ci = r % CIN;
        int fi = fo * STRIDE + df - 1;          // pad_f = 1 for all stages
        if (fi >= 0 && fi < FIN) {
          size_t b = b0 + nb;
          if (dt == 0)
            v = cachein[((size_t)b * CIN + ci) * FIN + fi];
          else if (XNHWC)
            v = xin[((size_t)b * FIN + fi) * CIN + ci];   // raw x: [B,F,1,2]
          else
            v = xin[((size_t)b * CIN + ci) * FIN + fi];   // NCHW activations
        }
      }
      Bh[idx] = (_Float16)v;
    }
    __syncthreads();

    // ---- WMMA main loop ----
    v8f acc[MT];
    #pragma unroll
    for (int mt = 0; mt < MT; ++mt)
      #pragma unroll
      for (int i = 0; i < 8; ++i) acc[mt][i] = 0.f;

    const int nb0 = wv * 16;
    #pragma unroll
    for (int ks = 0; ks < KSTEPS; ++ks) {
      // B fragment (K x 16): lane l15 = column n, k in {kb..kb+7, kb+16..kb+23}
      const int kb = ks * 32 + (lhi ? 8 : 0);
      const _Float16* bcol = Bh + (size_t)(nb0 + l15) * KPAD;
      v8h blo = *(const v8h*)(bcol + kb);
      v8h bhi = *(const v8h*)(bcol + kb + 16);
      v16h bfrag = __builtin_shufflevector(blo, bhi,
          0,1,2,3,4,5,6,7,8,9,10,11,12,13,14,15);
      // A fragments (16 x K): lane l15 = row m, k in {ka..ka+7, ka+16..ka+23}
      const int ka = ks * 32 + (lhi ? 8 : 0);
      #pragma unroll
      for (int mt = 0; mt < MT; ++mt) {
        const _Float16* wrow = Wh + (size_t)(mt * 16 + l15) * KPAD;
        v8h lo = *(const v8h*)(wrow + ka);
        v8h hi = *(const v8h*)(wrow + ka + 16);
        v16h afrag = __builtin_shufflevector(lo, hi,
            0,1,2,3,4,5,6,7,8,9,10,11,12,13,14,15);
        acc[mt] = __builtin_amdgcn_wmma_f32_16x16x32_f16(
            /*neg_a=*/false, afrag, /*neg_b=*/false, bfrag,
            /*c_mod=*/(short)0, acc[mt], /*reuse_a=*/false, /*reuse_b=*/false);
      }
    }

    // ---- epilogue: BN + PReLU, write x_i (and c_i copy) in NCHW ----
    const size_t bb = b0 + nb0 + l15;   // D layout: lane l15 = column n = batch
    #pragma unroll
    for (int mt = 0; mt < MT; ++mt) {
      #pragma unroll
      for (int i = 0; i < 8; ++i) {
        int co = mt * 16 + i + 8 * lhi; // D layout: vgpr i -> M = i (+8 hi half)
        float y = acc[mt][i] * sscale[co] + sshift[co];
        y = (y >= 0.f) ? y : y * salpha[co];
        size_t o = ((size_t)bb * COUT + co) * FOUT + fo;
        out0[o] = y;
        if (out1) out1[o] = y;
      }
    }
    __syncthreads();  // protect Bh before next fill
  }
}

// c0 = transpose of raw input: c0[b,c,0,f] = x[b,f,0,c]
__global__ __launch_bounds__(256)
void c0_transpose_kernel(const float* __restrict__ x, float* __restrict__ c0)
{
  size_t i = (size_t)blockIdx.x * 256 + threadIdx.x;   // over B*F = 8192*256
  size_t b = i >> 8, f = i & 255;
  const float* xs = x + (b * 256 + f) * 2;
  c0[(b * 2 + 0) * 256 + f] = xs[0];
  c0[(b * 2 + 1) * 256 + f] = xs[1];
}

static constexpr size_t smem_bytes(int cin, int cout, int kf) {
  int ktot = kf * 2 * cin;
  int kpad = ((ktot + 31) / 32) * 32;
  return (size_t)(cout + NB) * kpad * 2 + (size_t)3 * cout * 4;
}

extern "C" void kernel_launch(void* const* d_in, const int* in_sizes, int n_in,
                              void* d_out, int out_size, void* d_ws, size_t ws_size,
                              hipStream_t stream) {
  (void)in_sizes; (void)n_in; (void)out_size; (void)d_ws; (void)ws_size;

  const float* x      = (const float*)d_in[0];
  const float* cache  = (const float*)d_in[1];
  const float* cache1 = (const float*)d_in[2];
  const float* cache2 = (const float*)d_in[3];
  const float* cache3 = (const float*)d_in[4];
  const float* cache4 = (const float*)d_in[5];
  // params: 5 stages x {W, b, gamma, beta, mean, var, alpha} in dict order
  const float* const* P = (const float* const*)(d_in + 6);

  float* out = (float*)d_out;

  // output regions, flat in return order (x1,x2,x3,x4,x5,c0,c1,c2,c3,c4)
  const size_t S_X1 = (size_t)8192 * 32 * 127;
  const size_t S_X2 = (size_t)8192 * 32 * 64;
  const size_t S_X3 = (size_t)8192 * 32 * 64;
  const size_t S_X4 = (size_t)8192 * 64 * 64;
  const size_t S_X5 = (size_t)8192 * 128 * 64;
  const size_t S_C0 = (size_t)8192 * 2 * 256;
  const size_t OFF_X1 = 0;
  const size_t OFF_X2 = OFF_X1 + S_X1;
  const size_t OFF_X3 = OFF_X2 + S_X2;
  const size_t OFF_X4 = OFF_X3 + S_X3;
  const size_t OFF_X5 = OFF_X4 + S_X4;
  const size_t OFF_C0 = OFF_X5 + S_X5;
  const size_t OFF_C1 = OFF_C0 + S_C0;
  const size_t OFF_C2 = OFF_C1 + S_X1;
  const size_t OFF_C3 = OFF_C2 + S_X2;
  const size_t OFF_C4 = OFF_C3 + S_X3;

  // opt in to large dynamic LDS for the big stages (no-op if already allowed)
  (void)hipFuncSetAttribute((const void*)stage_kernel<64,128,64,64,1,3,false>,
                            hipFuncAttributeMaxDynamicSharedMemorySize,
                            (int)smem_bytes(64, 128, 3));
  (void)hipFuncSetAttribute((const void*)stage_kernel<32,64,64,64,1,3,false>,
                            hipFuncAttributeMaxDynamicSharedMemorySize,
                            (int)smem_bytes(32, 64, 3));

  const dim3 blk(TPB, 1, 1);
  const int GX = 8192 / NB;  // 128 batch tiles

  // c0 (pure layout transform of the raw input)
  c0_transpose_kernel<<<dim3((8192 * 256) / 256), dim3(256), 0, stream>>>(
      x, out + OFF_C0);

  // stage 1: 2->32, kf=5, stride 2, F 256->127
  stage_kernel<2, 32, 256, 127, 2, 5, true>
      <<<dim3(GX, (127 + FO_PER_WG - 1) / FO_PER_WG), blk,
         smem_bytes(2, 32, 5), stream>>>(
      x, cache, P[0], P[1], P[2], P[3], P[4], P[5], P[6],
      out + OFF_X1, out + OFF_C1);

  // stage 2: 32->32, kf=3, stride 2, F 127->64
  stage_kernel<32, 32, 127, 64, 2, 3, false>
      <<<dim3(GX, 64 / FO_PER_WG), blk, smem_bytes(32, 32, 3), stream>>>(
      out + OFF_X1, cache1, P[7], P[8], P[9], P[10], P[11], P[12], P[13],
      out + OFF_X2, out + OFF_C2);

  // stage 3: 32->32, kf=3, stride 1, F 64
  stage_kernel<32, 32, 64, 64, 1, 3, false>
      <<<dim3(GX, 64 / FO_PER_WG), blk, smem_bytes(32, 32, 3), stream>>>(
      out + OFF_X2, cache2, P[14], P[15], P[16], P[17], P[18], P[19], P[20],
      out + OFF_X3, out + OFF_C3);

  // stage 4: 32->64, kf=3, stride 1, F 64
  stage_kernel<32, 64, 64, 64, 1, 3, false>
      <<<dim3(GX, 64 / FO_PER_WG), blk, smem_bytes(32, 64, 3), stream>>>(
      out + OFF_X3, cache3, P[21], P[22], P[23], P[24], P[25], P[26], P[27],
      out + OFF_X4, out + OFF_C4);

  // stage 5: 64->128, kf=3, stride 1, F 64 (no cache output)
  stage_kernel<64, 128, 64, 64, 1, 3, false>
      <<<dim3(GX, 64 / FO_PER_WG), blk, smem_bytes(64, 128, 3), stream>>>(
      out + OFF_X4, cache4, P[28], P[29], P[30], P[31], P[32], P[33], P[34],
      out + OFF_X5, nullptr);
}
